// PointNetSetAbstraction5_53154515256139
// MI455X (gfx1250) — matrix operand order repaired and checked by reading, not compiled, not run on hardware
//
#include <hip/hip_runtime.h>
#include <hip/hip_bf16.h>
#include <stdint.h>

typedef __attribute__((ext_vector_type(16))) _Float16 v16h;
typedef __attribute__((ext_vector_type(8)))  _Float16 v8h;
typedef __attribute__((ext_vector_type(2)))  _Float16 h2;
typedef __attribute__((ext_vector_type(8)))  float    v8f;

#define B_    16
#define N_    4096
#define S_    1024
#define K_    32
#define CPAD  32
#define P_    (B_*S_*K_)      /* 524288 pixels */
#define R2_   0.04f
#define EPS_  1e-5f

/* ---- workspace layout (bytes) ---- */
#define WS_FPSIDX   (0u)                  /* 16384 ints            */
#define WS_NEWXYZ   (64u*1024u)           /* 49152 floats          */
#define WS_STATS    (256u*1024u)          /* 384 floats            */
#define WS_BN       (258u*1024u)          /* 384 floats            */
#define WS_APACK    (264u*1024u)          /* packed W frags, 8 KB  */
#define WS_ZMAX     (280u*1024u)          /* 1048576 uints (4 MB)  */
#define WS_X0       (8u*1024u*1024u)      /* P*32 halfs (32 MB)    */
#define WS_Z0       (40u*1024u*1024u)     /* 32 MB                 */
#define WS_Z1       (72u*1024u*1024u)     /* 32 MB                 */

/* Apack offsets in halfs: layer0 2 m-tiles, layer1 2, layer2 4; 16 halfs/lane */
#define AP0  0
#define AP1  1024
#define AP2  2048

__device__ __forceinline__ unsigned fenc(float f) {
    unsigned u = __float_as_uint(f);
    return (u & 0x80000000u) ? ~u : (u | 0x80000000u);
}
__device__ __forceinline__ float fdec(unsigned u) {
    return __uint_as_float((u & 0x80000000u) ? (u & 0x7fffffffu) : ~u);
}

/* -------- init: zero the 384 stat accumulators -------- */
__global__ __launch_bounds__(256) void init_ws_kernel(float* stats) {
    int i = blockIdx.x * 256 + threadIdx.x;
    if (i < 384) stats[i] = 0.f;
}

/* -------- pack weights into per-lane WMMA A-fragment layout (f16, zero-padded) --------
 * entry t = mt*32 + lane; 16 halfs each; contiguous k order used here (same as B):
 * low lanes hold K{0..15}, high lanes K{16..31}.                                       */
__global__ __launch_bounds__(256)
void pack_w_kernel(const float* __restrict__ W, _Float16* __restrict__ Apack,
                   int O, int CIN) {
    int t = threadIdx.x;
    if (t >= (O / 16) * 32) return;
    int mt = t >> 5, lane = t & 31;
    int khi = lane >> 4;
    int orow = mt * 16 + (lane & 15);
    v16h a;
#pragma unroll
    for (int i = 0; i < 8; i++) {
        int kb = 2 * i + khi * 16;
        float w0 = (kb < CIN) ? W[orow * CIN + kb] : 0.f;
        float w1 = (kb + 1 < CIN) ? W[orow * CIN + kb + 1] : 0.f;
        a[2 * i] = (_Float16)w0;
        a[2 * i + 1] = (_Float16)w1;
    }
    *(v16h*)(Apack + (size_t)t * 16) = a;
}

/* -------- farthest point sampling: one block per batch -------- */
__global__ __launch_bounds__(256)
void fps_kernel(const float* __restrict__ xyz, int* __restrict__ fpsIdx,
                float* __restrict__ newXyz, float* __restrict__ outXyz) {
    int b = blockIdx.x, tid = threadIdx.x;
    const float* xb = xyz + (size_t)b * 3 * N_;
    float px[16], py[16], pz[16], dist[16];
#pragma unroll
    for (int j = 0; j < 16; j++) {
        int n = j * 256 + tid;
        px[j] = xb[n]; py[j] = xb[N_ + n]; pz[j] = xb[2 * N_ + n];
        dist[j] = 1e10f;
    }
    __shared__ float sVal[8];
    __shared__ int   sIdx[8];
    __shared__ int   sBest;
    int lane = tid & 31, wv = tid >> 5;
    int farthest = 0;
    for (int s = 0; s < S_; s++) {
        if (tid == 0) {
            fpsIdx[b * S_ + s] = farthest;
            float cx = xb[farthest], cy = xb[N_ + farthest], cz = xb[2 * N_ + farthest];
            float* nx = newXyz + ((size_t)b * S_ + s) * 3;
            nx[0] = cx; nx[1] = cy; nx[2] = cz;
            outXyz[((size_t)b * 3 + 0) * S_ + s] = cx;
            outXyz[((size_t)b * 3 + 1) * S_ + s] = cy;
            outXyz[((size_t)b * 3 + 2) * S_ + s] = cz;
        }
        float cx = xb[farthest], cy = xb[N_ + farthest], cz = xb[2 * N_ + farthest];
        float bv = -1.f; int bi = 0;
#pragma unroll
        for (int j = 0; j < 16; j++) {
            float dx = px[j] - cx, dy = py[j] - cy, dz = pz[j] - cz;
            float d = dx * dx + dy * dy + dz * dz;
            dist[j] = fminf(dist[j], d);
            if (dist[j] > bv) { bv = dist[j]; bi = j * 256 + tid; }
        }
#pragma unroll
        for (int off = 16; off > 0; off >>= 1) {
            float ov = __shfl_down(bv, off, 32);
            int   oi = __shfl_down(bi, off, 32);
            if (ov > bv || (ov == bv && oi < bi)) { bv = ov; bi = oi; }
        }
        if (lane == 0) { sVal[wv] = bv; sIdx[wv] = bi; }
        __syncthreads();
        if (tid == 0) {
            float best = sVal[0]; int besti = sIdx[0];
#pragma unroll
            for (int w = 1; w < 8; w++)
                if (sVal[w] > best || (sVal[w] == best && sIdx[w] < besti)) { best = sVal[w]; besti = sIdx[w]; }
            sBest = besti;
        }
        __syncthreads();
        farthest = sBest;
    }
}

/* -------- ball query + grouping: one wave per (b,s), writes f16 feature rows -------- */
__device__ __forceinline__ void write_row(_Float16* __restrict__ X0, size_t p,
                                          float f0, float f1, float f2,
                                          const float* __restrict__ pb, int n) {
    _Float16 row[CPAD];
    row[0] = (_Float16)f0; row[1] = (_Float16)f1; row[2] = (_Float16)f2;
#pragma unroll
    for (int c = 0; c < 6; c++) row[3 + c] = (_Float16)pb[c * N_ + n];
#pragma unroll
    for (int c = 9; c < CPAD; c++) row[c] = (_Float16)0.f;
    v8h* dst = (v8h*)(X0 + p * CPAD);
#pragma unroll
    for (int q = 0; q < 4; q++) {
        v8h t;
#pragma unroll
        for (int e = 0; e < 8; e++) t[e] = row[q * 8 + e];
        dst[q] = t;
    }
}

__global__ __launch_bounds__(256)
void group_kernel(const float* __restrict__ xyz, const float* __restrict__ pts,
                  const float* __restrict__ newXyz, _Float16* __restrict__ X0) {
    int lane = threadIdx.x & 31, wv = threadIdx.x >> 5;
    int g = blockIdx.x * 8 + wv;          /* b*S + s */
    int b = g >> 10;
    const float* xb = xyz + (size_t)b * 3 * N_;
    const float* pb = pts + (size_t)b * 6 * N_;
    float qx = newXyz[(size_t)g * 3 + 0];
    float qy = newXyz[(size_t)g * 3 + 1];
    float qz = newXyz[(size_t)g * 3 + 2];
    int count = 0, first = -1;
    for (int base = 0; base < N_ && count < K_; base += 32) {
        int n = base + lane;
        float x = xb[n], y = xb[N_ + n], z = xb[2 * N_ + n];
        float dx = x - qx, dy = y - qy, dz = z - qz;
        float d = dx * dx + dy * dy + dz * dz;
        bool in = (d <= R2_);
        unsigned bal = (unsigned)__ballot(in);
        if (first < 0 && bal) first = base + __ffs(bal) - 1;
        int slot = count + __popc(bal & ((1u << lane) - 1u));
        if (in && slot < K_)
            write_row(X0, (size_t)g * K_ + slot, dx, dy, dz, pb, n);
        count += __popc(bal);
    }
    if (count < K_) {                       /* pad with first in-radius point */
        if (first < 0) first = 0;
        int n = first;
        float x = xb[n], y = xb[N_ + n], z = xb[2 * N_ + n];
        int slot = count + lane;
        if (slot < K_)
            write_row(X0, (size_t)g * K_ + slot, x - qx, y - qy, z - qz, pb, n);
    }
}

/* -------- WMMA GEMM + bias + (optional BN-in) + stats + (store | fused k-max) --------
 * D[o,p] = sum_c W[o,c]*act(x[c,p]) + bias[o] using v_wmma_f32_16x16x32_f16.
 * One wave per 16-pixel tile; loops over 16-channel M tiles. A-fragments are
 * pre-packed (L2-resident, one 32B load). Stats via fire-and-forget ds_add_f32.
 * FUSED: each block covers exactly 4 complete (b,s) groups, so the k-max is
 * block-local: ds_max_u32 into LDS, then a plain coalesced store (no atomics). */
template<int O, bool BN_IN, bool FUSED>
__global__ __launch_bounds__(256)
void gemm_bn_kernel(const _Float16* __restrict__ Xin,
                    const _Float16* __restrict__ Apack,
                    const float* __restrict__ bias,
                    const float* __restrict__ bnMulIn, const float* __restrict__ bnAddIn,
                    _Float16* __restrict__ Zout, unsigned* __restrict__ zmax,
                    float* __restrict__ gsum, float* __restrict__ gsumsq) {
    __shared__ float    sSum[64], sSq[64];
    __shared__ float    sMul[32], sAdd[32];
    __shared__ unsigned sZmax[256];        /* 4 local s-groups x 64 channels */
    int tid = threadIdx.x;
    if (tid < 64) { sSum[tid] = 0.f; sSq[tid] = 0.f; }
    if (BN_IN && tid < 32) { sMul[tid] = bnMulIn[tid]; sAdd[tid] = bnAddIn[tid]; }
    if (FUSED) sZmax[tid] = fenc(-3.0e38f);
    __syncthreads();

    int lane = tid & 31, wv = tid >> 5;
    int tile = blockIdx.x * 8 + wv;       /* pixel tile (16 pixels) */
    int p0 = tile * 16;
    int nn = lane & 15;
    int khi = lane >> 4;
    int p = p0 + nn;
    int localS = (wv * 16 + nn) >> 5;     /* which of the block's 4 (b,s) groups */

    /* B fragment: 32x16 f16; lanes 0-15 hold K=0..15, lanes 16-31 hold K=16..31 */
    v16h bfrag;
    {
        const _Float16* row = Xin + (size_t)p * CPAD + khi * 16;
#pragma unroll
        for (int i = 0; i < 8; i++) {
            h2 h = *(const h2*)(row + 2 * i);
            if (BN_IN) {
                int k = khi * 16 + 2 * i;
                float y0 = fmaxf(0.f, (float)h[0] * sMul[k] + sAdd[k]);
                float y1 = fmaxf(0.f, (float)h[1] * sMul[k + 1] + sAdd[k + 1]);
                h[0] = (_Float16)y0; h[1] = (_Float16)y1;
            }
            bfrag[2 * i] = h[0]; bfrag[2 * i + 1] = h[1];
        }
    }

#pragma unroll
    for (int mt = 0; mt < O / 16; ++mt) {
        int m0 = mt * 16;
        /* A fragment: single 32B load from pre-packed layout */
        v16h afrag = *(const v16h*)(Apack + ((size_t)(mt * 32 + lane)) * 16);

        /* C/D layout: VGPR r -> M = r + 8*khi, N = lane&15; fold bias into C */
        v8f acc;
#pragma unroll
        for (int r = 0; r < 8; r++) acc[r] = bias[m0 + khi * 8 + r];

        acc = __builtin_amdgcn_wmma_f32_16x16x32_f16(false, afrag, false, bfrag,
                                                     (short)0, acc, false, false);

        if (!FUSED) {
            v8h zh;
#pragma unroll
            for (int r = 0; r < 8; r++) zh[r] = (_Float16)acc[r];
            *(v8h*)(Zout + (size_t)p * O + m0 + khi * 8) = zh;
        }

        /* per-lane fire-and-forget LDS atomics; the DS unit does the reduction */
#pragma unroll
        for (int r = 0; r < 8; r++) {
            float v = acc[r];
            int o = m0 + khi * 8 + r;
            atomicAdd(&sSum[o], v);
            atomicAdd(&sSq[o], v * v);
            if (FUSED) atomicMax(&sZmax[localS * 64 + o], fenc(v));
        }
    }
    __syncthreads();
    if (tid < O) {
        atomicAdd(&gsum[tid], sSum[tid]);
        atomicAdd(&gsumsq[tid], sSq[tid]);
    }
    /* each (b,s,o) is owned by exactly this block: plain coalesced store */
    if (FUSED) zmax[(size_t)blockIdx.x * 256 + tid] = sZmax[tid];
}

/* -------- per-layer BN finalize: mul = g*rsqrt(var+eps); add = be - mean*mul -------- */
__global__ void bn_finalize_kernel(const float* gsum, const float* gsumsq,
                                   const float* gamma, const float* beta,
                                   float* mul, float* add, int O) {
    int o = threadIdx.x;
    if (o < O) {
        float m = gsum[o] * (1.0f / (float)P_);
        float var = gsumsq[o] * (1.0f / (float)P_) - m * m;
        float sc = gamma[o] * rsqrtf(var + EPS_);
        mul[o] = sc;
        add[o] = beta[o] - m * sc;
    }
}

/* -------- final: BN2+ReLU on the pre-BN k-max, write new_points -------- */
__global__ __launch_bounds__(256)
void final_out_kernel(const unsigned* __restrict__ zmax, const float* __restrict__ mul2,
                      const float* __restrict__ add2, float* __restrict__ out) {
    int j = blockIdx.x * 256 + threadIdx.x;       /* over B*64*S, output-ordered */
    if (j >= B_ * 64 * S_) return;
    int s = j & (S_ - 1);
    int ob = j >> 10;
    int o = ob & 63;
    int b = ob >> 6;
    float z = fdec(zmax[(((size_t)b << 10) + s) * 64 + o]);
    float y = fmaxf(0.f, z * mul2[o] + add2[o]);
    out[B_ * 3 * S_ + j] = y;
}

extern "C" void kernel_launch(void* const* d_in, const int* in_sizes, int n_in,
                              void* d_out, int out_size, void* d_ws, size_t ws_size,
                              hipStream_t stream) {
    const float* xyz = (const float*)d_in[0];
    const float* pts = (const float*)d_in[1];
    const float* W0 = (const float*)d_in[2];  const float* b0 = (const float*)d_in[3];
    const float* g0 = (const float*)d_in[4];  const float* be0 = (const float*)d_in[5];
    const float* W1 = (const float*)d_in[6];  const float* b1 = (const float*)d_in[7];
    const float* g1 = (const float*)d_in[8];  const float* be1 = (const float*)d_in[9];
    const float* W2 = (const float*)d_in[10]; const float* b2 = (const float*)d_in[11];
    const float* g2 = (const float*)d_in[12]; const float* be2 = (const float*)d_in[13];

    char* ws = (char*)d_ws;
    int*       fpsIdx = (int*)(ws + WS_FPSIDX);
    float*     newXyz = (float*)(ws + WS_NEWXYZ);
    float*     stats  = (float*)(ws + WS_STATS);
    float*     bn     = (float*)(ws + WS_BN);
    _Float16*  Apack  = (_Float16*)(ws + WS_APACK);
    unsigned*  zmax   = (unsigned*)(ws + WS_ZMAX);
    _Float16*  X0     = (_Float16*)(ws + WS_X0);
    _Float16*  Z0     = (_Float16*)(ws + WS_Z0);
    _Float16*  Z1     = (_Float16*)(ws + WS_Z1);
    float*     out    = (float*)d_out;

    init_ws_kernel<<<2, 256, 0, stream>>>(stats);
    pack_w_kernel<<<1, 256, 0, stream>>>(W0, Apack + AP0, 32, 9);
    pack_w_kernel<<<1, 256, 0, stream>>>(W1, Apack + AP1, 32, 32);
    pack_w_kernel<<<1, 256, 0, stream>>>(W2, Apack + AP2, 64, 32);

    fps_kernel<<<B_, 256, 0, stream>>>(xyz, fpsIdx, newXyz, out);
    group_kernel<<<(B_ * S_) / 8, 256, 0, stream>>>(xyz, pts, newXyz, X0);

    /* layer 0: 9->32 (zero-padded to K=32), no input BN */
    gemm_bn_kernel<32, false, false><<<4096, 256, 0, stream>>>(
        X0, Apack + AP0, b0, nullptr, nullptr, Z0, nullptr, stats + 0, stats + 64);
    bn_finalize_kernel<<<1, 64, 0, stream>>>(stats + 0, stats + 64, g0, be0, bn + 0, bn + 64, 32);

    /* layer 1: 32->32, applies BN0+ReLU on load */
    gemm_bn_kernel<32, true, false><<<4096, 256, 0, stream>>>(
        Z0, Apack + AP1, b1, bn + 0, bn + 64, Z1, nullptr, stats + 128, stats + 192);
    bn_finalize_kernel<<<1, 64, 0, stream>>>(stats + 128, stats + 192, g1, be1, bn + 128, bn + 192, 32);

    /* layer 2: 32->64, applies BN1+ReLU on load, fused k-max (no Z2 materialization) */
    gemm_bn_kernel<64, true, true><<<4096, 256, 0, stream>>>(
        Z1, Apack + AP2, b2, bn + 128, bn + 192, nullptr, zmax, stats + 256, stats + 320);
    bn_finalize_kernel<<<1, 64, 0, stream>>>(stats + 256, stats + 320, g2, be2, bn + 256, bn + 320, 64);

    final_out_kernel<<<4096, 256, 0, stream>>>(zmax, bn + 256, bn + 320, out);
}
